// QLSTM_65481071404400
// MI455X (gfx1250) — matrix operand order repaired and checked by reading, not compiled, not run on hardware
//
#include <hip/hip_runtime.h>

// ---------------------------------------------------------------------------
// QLSTM for MI455X (gfx1250): weight-norm + fused 4-gate LSTM.
//   T=512, B=64, D=512, H=512.  z = [x,h] @ W_all^T + b, gates f,i,u,o.
//   phase0 : weight-norm -> Wx f16 [2048][512] row-major, WhT f16 [2048][512].
//   phase0b: cast inputs f32 -> f16.
//   phase1 : Zx = x @ Wx^T + b. Per block: 8 waves x 8 M-tiles share Wx
//            N-tiles staged into LDS via global_load_async_to_lds_b128
//            (double-buffered, ASYNCcnt), B-frags via ds_load_b128, math via
//            v_wmma_f32_16x16x32_f16 with register-resident A-frags.
//   phase2 : persistent 32-block kernel; Wh B-frags live in VGPRs for all 512
//            steps; next-step Zx prefetched (global_prefetch_b8); grid
//            counter barrier per step; h double-buffered in global (L2).
// ---------------------------------------------------------------------------

typedef __attribute__((ext_vector_type(16))) _Float16 v16h;
typedef __attribute__((ext_vector_type(8)))  _Float16 v8h;
typedef __attribute__((ext_vector_type(4)))  _Float16 v4h;
typedef __attribute__((ext_vector_type(8)))  float    v8f;

#define T_STEPS 512
#define BATCH   64
#define DDIM    512
#define HDIM    512
#define NOUT    2048   // 4*H
#define KIN     1024   // D+H
#define BPAD    520    // padded f16 row stride for LDS B tiles

static __device__ __forceinline__ v16h load16(const _Float16* lo, const _Float16* hi) {
  v16h r;
  *reinterpret_cast<v8h*>(&r)       = *reinterpret_cast<const v8h*>(lo);
  *(reinterpret_cast<v8h*>(&r) + 1) = *reinterpret_cast<const v8h*>(hi);
  return r;
}

static __device__ __forceinline__ float sigmoidf_(float x) {
  return 1.0f / (1.0f + __expf(-x));
}

// LDS byte offset of a generic pointer known to point into LDS
static __device__ __forceinline__ unsigned lds_off(const void* p) {
  return (unsigned)(unsigned long long)
      (const __attribute__((address_space(3))) void*)p;
}

// async DMA: 16 bytes global -> LDS, tracked by ASYNCcnt (per-lane addresses)
static __device__ __forceinline__ void async_g2l_b128(unsigned lds_byte,
                                                      const void* gaddr) {
  asm volatile("global_load_async_to_lds_b128 %0, %1, off"
               :: "v"(lds_byte), "v"(gaddr) : "memory");
}

// ---------------- phase 0: weight norm -> f16 Wx, WhT, bias ----------------
__global__ void qlstm_prep_weights(const float* __restrict__ v,
                                   const float* __restrict__ g,
                                   const float* __restrict__ b,
                                   _Float16* __restrict__ wx,
                                   _Float16* __restrict__ wht,
                                   float* __restrict__ ball) {
  __shared__ float red[256];
  const int o = blockIdx.x;                 // 0..2047 output row
  const float* row = v + (size_t)o * KIN;
  float s = 0.0f;
  for (int k = threadIdx.x; k < KIN; k += 256) { float t = row[k]; s += t * t; }
  red[threadIdx.x] = s;
  __syncthreads();
  for (int off = 128; off > 0; off >>= 1) {
    if (threadIdx.x < off) red[threadIdx.x] += red[threadIdx.x + off];
    __syncthreads();
  }
  const float scale = g[o] / sqrtf(red[0]);
  for (int k = threadIdx.x; k < DDIM; k += 256)
    wx[(size_t)o * DDIM + k] = (_Float16)(row[k] * scale);
  for (int k = threadIdx.x; k < HDIM; k += 256)
    wht[(size_t)o * HDIM + k] = (_Float16)(row[DDIM + k] * scale);
  if (threadIdx.x == 0) ball[o] = b[o];
}

// ---------------- phase 0b: cast inputs f32 -> f16 -------------------------
__global__ void qlstm_cast_inputs(const float* __restrict__ x,
                                  _Float16* __restrict__ xh) {
  const int i = (blockIdx.x * 256 + threadIdx.x) * 4;   // n divisible by 1024
  const float4 vv = *reinterpret_cast<const float4*>(x + i);
  v4h o;
  o[0] = (_Float16)vv.x; o[1] = (_Float16)vv.y;
  o[2] = (_Float16)vv.z; o[3] = (_Float16)vv.w;
  *reinterpret_cast<v4h*>(xh + i) = o;
}

// ---------------- phase 1: Zx = x @ Wx^T + b  (parallel over T*B) ----------
// grid: 256 blocks x 256 threads (8 waves). wave -> one M-tile (16 rows).
// All waves share the current Wx N-tile, staged in LDS by async DMA with
// double buffering: copy nt+1 while computing nt.
__global__ void __launch_bounds__(256)
qlstm_gemm_x(const _Float16* __restrict__ xh,
             const _Float16* __restrict__ wx,
             const float* __restrict__ ball,
             float* __restrict__ zx) {
  __shared__ _Float16 Bstage[2][16][BPAD];   // 2 x 16 rows x 520 f16 = 33 KB

  const int tid   = threadIdx.x;
  const int wave  = tid >> 5;
  const int lane  = tid & 31;
  const int l15   = lane & 15;
  const int hi    = lane >> 4;
  const int mtile = blockIdx.x * 8 + wave;            // 0..2047

  // register-resident A-frags for this wave's 16 rows (K = 512)
  const _Float16* arow = xh + (size_t)(mtile * 16 + l15) * DDIM;
  v16h A[16];
#pragma unroll
  for (int k = 0; k < 16; ++k) {
    const int K0 = k * 32;
    A[k] = load16(arow + K0 + hi * 16, arow + K0 + hi * 16 + 8);
  }

  // async-copy helper: tile nt (16 rows x 512 f16 = 1024 x 16B chunks),
  // 4 chunks per thread per tile -> 4 ASYNCcnt per wave per tile
  const unsigned base0 = lds_off(&Bstage[0][0][0]);
  const unsigned base1 = lds_off(&Bstage[1][0][0]);

  // prologue: stage nt = 0 into buffer 0
  {
#pragma unroll
    for (int i = 0; i < 4; ++i) {
      const int c   = tid * 4 + i;          // 0..1023
      const int row = c >> 6, ch = c & 63;
      async_g2l_b128(base0 + (unsigned)(row * (BPAD * 2) + ch * 16),
                     wx + (size_t)row * DDIM + ch * 8);
    }
  }

  for (int nt = 0; nt < 128; ++nt) {
    const int cur = nt & 1;
    // stage nt+1 into the other buffer while we compute nt
    if (nt + 1 < 128) {
      const unsigned nbase = (cur ? base0 : base1);
      const _Float16* gsrc = wx + (size_t)(nt + 1) * 16 * DDIM;
#pragma unroll
      for (int i = 0; i < 4; ++i) {
        const int c   = tid * 4 + i;
        const int row = c >> 6, ch = c & 63;
        async_g2l_b128(nbase + (unsigned)(row * (BPAD * 2) + ch * 16),
                       gsrc + (size_t)row * DDIM + ch * 8);
      }
      asm volatile("s_wait_asynccnt 0x4" ::: "memory");  // nt's copies done
    } else {
      asm volatile("s_wait_asynccnt 0x0" ::: "memory");
    }
    __syncthreads();                                     // tile nt visible

    const float bias = ball[nt * 16 + l15];
    v8f acc = {bias, bias, bias, bias, bias, bias, bias, bias};
    const _Float16* brow = &Bstage[cur][l15][0];
#pragma unroll
    for (int k = 0; k < 16; ++k) {
      const int K0 = k * 32;
      v16h Bf = load16(brow + K0 + hi * 16, brow + K0 + hi * 16 + 8);
      acc = __builtin_amdgcn_wmma_f32_16x16x32_f16(false, A[k], false, Bf,
                                                   (short)0, acc, false, false);
    }
    __syncthreads();   // everyone done reading buf[cur] before nt+2 refills it

    float* zb = zx + (size_t)(mtile * 16 + hi * 8) * NOUT + nt * 16 + l15;
#pragma unroll
    for (int r = 0; r < 8; ++r) zb[(size_t)r * NOUT] = acc[r];
  }
}

// ---------------- phase 2: persistent recurrent kernel ---------------------
// 32 blocks x 512 threads (16 waves). Block blk owns h columns
// [16*blk, 16*blk+16) across all 4 gates. Wave w: gate=w>>2, mtile=w&3.
// Wh B-frags (16 x v16h = 128 VGPRs) persist across all 512 steps.
__global__ void __launch_bounds__(512)
qlstm_recurrent(const _Float16* __restrict__ wht,
                const float* __restrict__ zx,
                _Float16* __restrict__ hbuf,     // [2][64][512] f16, zeroed
                float* __restrict__ out,         // outputs ++ hx ++ cx
                unsigned* __restrict__ syncc) {  // zeroed counter
  __shared__ float z_lds[BATCH][68];   // [b][gate*16+j], padded
  __shared__ float c_lds[BATCH][17];   // cell state for this block's 16 cols

  const int blk  = blockIdx.x;         // 0..31
  const int tid  = threadIdx.x;
  const int wave = tid >> 5;
  const int lane = tid & 31;
  const int gate = wave >> 2;          // 0..3 (f,i,u,o)
  const int mt   = wave & 3;           // batch tile 0..3
  const int l15  = lane & 15;
  const int hi   = lane >> 4;

  // persistent B-frags for this wave's output column group
  const int nglob = gate * HDIM + blk * 16 + l15;
  const _Float16* brow = wht + (size_t)nglob * HDIM;
  v16h Bf[16];
#pragma unroll
  for (int k = 0; k < 16; ++k)
    Bf[k] = load16(brow + k * 32 + hi * 16, brow + k * 32 + hi * 16 + 8);

  for (int i = tid; i < BATCH * 17; i += 512) (&c_lds[0][0])[i] = 0.0f;
  __syncthreads();

  const size_t zoff = (size_t)(mt * 16 + hi * 8) * NOUT + nglob;

  for (int t = 0; t < T_STEPS; ++t) {
    const int p = t & 1;
    const _Float16* hprev = hbuf + (size_t)p * BATCH * HDIM;

    // C-init = Zx tile (bias already folded in by phase 1)
    v8f acc;
    {
      const float* zb = zx + (size_t)t * BATCH * NOUT + zoff;
#pragma unroll
      for (int r = 0; r < 8; ++r) acc[r] = zb[(size_t)r * NOUT];
    }
    // pull next step's Zx tile toward this WGP while we do the WMMAs
    if (t + 1 < T_STEPS) {
      const float* zp = zx + (size_t)(t + 1) * BATCH * NOUT + zoff;
#pragma unroll
      for (int r = 0; r < 8; ++r)
        __builtin_prefetch(zp + (size_t)r * NOUT, 0, 3);
    }

    const _Float16* arow = hprev + (size_t)(mt * 16 + l15) * HDIM;
#pragma unroll
    for (int k = 0; k < 16; ++k) {
      const int K0 = k * 32;
      v16h A = load16(arow + K0 + hi * 16, arow + K0 + hi * 16 + 8);
      acc = __builtin_amdgcn_wmma_f32_16x16x32_f16(false, A, false, Bf[k],
                                                   (short)0, acc, false, false);
    }

#pragma unroll
    for (int r = 0; r < 8; ++r)
      z_lds[mt * 16 + hi * 8 + r][gate * 16 + l15] = acc[r];
    __syncthreads();

    // elementwise cell update: 64x16 = 1024 elems, 2 per thread
    _Float16* hnext = hbuf + (size_t)(1 - p) * BATCH * HDIM;
#pragma unroll
    for (int e = 0; e < 2; ++e) {
      const int idx = tid + e * 512;
      const int bb = idx >> 4, j = idx & 15;
      const float f  = sigmoidf_(z_lds[bb][j]);
      const float ii = sigmoidf_(z_lds[bb][16 + j]);
      const float u  = tanhf(z_lds[bb][32 + j]);
      const float o  = sigmoidf_(z_lds[bb][48 + j]);
      const float c  = f * c_lds[bb][j] + ii * u;
      c_lds[bb][j] = c;
      const float h = o * tanhf(c);
      const int col = blk * 16 + j;
      hnext[(size_t)bb * HDIM + col] = (_Float16)h;
      out[(size_t)t * BATCH * HDIM + (size_t)bb * HDIM + col] = h;
    }
    __syncthreads();

    // grid-wide split barrier (release h writes, acquire for next step)
    if (tid == 0) {
      __threadfence();
      __hip_atomic_fetch_add(syncc, 1u, __ATOMIC_RELEASE, __HIP_MEMORY_SCOPE_AGENT);
      const unsigned target = 32u * (unsigned)(t + 1);
      while (__hip_atomic_load(syncc, __ATOMIC_ACQUIRE, __HIP_MEMORY_SCOPE_AGENT) < target) {}
    }
    __syncthreads();
    __threadfence();   // every thread: invalidate stale h lines before reload
  }

  // tail: hx = h_T (copy from last outputs slice), cx from LDS
  float* hx  = out + (size_t)T_STEPS * BATCH * HDIM;
  float* cxo = hx + (size_t)BATCH * HDIM;
#pragma unroll
  for (int e = 0; e < 2; ++e) {
    const int idx = tid + e * 512;
    const int bb = idx >> 4, j = idx & 15;
    const int col = blk * 16 + j;
    hx[(size_t)bb * HDIM + col] =
        out[(size_t)(T_STEPS - 1) * BATCH * HDIM + (size_t)bb * HDIM + col];
    cxo[(size_t)bb * HDIM + col] = c_lds[bb][j];
  }
}

// ---------------------------------------------------------------------------
extern "C" void kernel_launch(void* const* d_in, const int* in_sizes, int n_in,
                              void* d_out, int out_size, void* d_ws, size_t ws_size,
                              hipStream_t stream) {
  (void)in_sizes; (void)n_in; (void)out_size; (void)ws_size;
  const float* x  = (const float*)d_in[0];   // [512,64,512]
  const float* v  = (const float*)d_in[1];   // [4,512,1024]
  const float* g  = (const float*)d_in[2];   // [4,512]
  const float* b  = (const float*)d_in[3];   // [4,512]
  float* out = (float*)d_out;

  char* ws = (char*)d_ws;
  unsigned* syncc = (unsigned*)ws;                               size_t off = 256;
  _Float16* xh   = (_Float16*)(ws + off);  off += (size_t)T_STEPS * BATCH * DDIM * 2; // 32 MB
  _Float16* wx   = (_Float16*)(ws + off);  off += (size_t)NOUT * DDIM * 2;            // 2 MB
  _Float16* wht  = (_Float16*)(ws + off);  off += (size_t)NOUT * HDIM * 2;            // 2 MB
  float*    ball = (float*)(ws + off);     off += (size_t)NOUT * 4;                   // 8 KB
  _Float16* hbuf = (_Float16*)(ws + off);  off += (size_t)2 * BATCH * HDIM * 2;       // 128 KB
  float*    zx   = (float*)(ws + off);     // 512*64*2048*4 = 256 MB

  hipMemsetAsync(syncc, 0, sizeof(unsigned), stream);
  hipMemsetAsync(hbuf, 0, (size_t)2 * BATCH * HDIM * 2, stream);

  qlstm_prep_weights<<<NOUT, 256, 0, stream>>>(v, g, b, wx, wht, ball);
  qlstm_cast_inputs<<<(T_STEPS * BATCH * DDIM) / 1024, 256, 0, stream>>>(x, xh);
  qlstm_gemm_x<<<256, 256, 0, stream>>>(xh, wx, ball, zx);
  qlstm_recurrent<<<32, 512, 0, stream>>>(wht, zx, hbuf, out, syncc);
}
